// MambaSequenceModel_40407052321101
// MI455X (gfx1250) — compile-verified, matrix operand order checked by reading, not run on hardware
//
#include <hip/hip_runtime.h>
#include <hip/hip_bf16.h>

// ---------------- Model constants ----------------
#define D_MODEL   128
#define N_LAYERS  4
#define D_STATE   16
#define D_CONV    4
#define D_INNER   256
#define DT_RANK   8
#define INPUT_DIM 64
#define NUM_CLS   2
#define BATCH     32
#define SEQ       1024
#define MROWS     (BATCH * SEQ)      // 32768
#define EPS       1e-5f

// ---------------- WMMA fragment types (gfx1250, wave32) ----------------
typedef __attribute__((ext_vector_type(16))) __bf16 v16bf;
typedef __attribute__((ext_vector_type(8)))  float  v8f;

// CDNA5 16-bit A/B fragment layout (ISA 7.12.2): lane = half*16 + r holds row r,
// element e maps to K(e) = 16*(e>>3) + 8*half + (e&7)  -> two contiguous 8-float groups.
__device__ __forceinline__ v16bf load_frag_f32(const float* __restrict__ p) {
    const float4* q0 = (const float4*)p;          // K group 0: 8 floats
    const float4* q1 = (const float4*)(p + 16);   // K group 1: 8 floats
    float4 x0 = q0[0], x1 = q0[1];
    float4 x2 = q1[0], x3 = q1[1];
    v16bf f;
    f[0]  = (__bf16)x0.x; f[1]  = (__bf16)x0.y; f[2]  = (__bf16)x0.z; f[3]  = (__bf16)x0.w;
    f[4]  = (__bf16)x1.x; f[5]  = (__bf16)x1.y; f[6]  = (__bf16)x1.z; f[7]  = (__bf16)x1.w;
    f[8]  = (__bf16)x2.x; f[9]  = (__bf16)x2.y; f[10] = (__bf16)x2.z; f[11] = (__bf16)x2.w;
    f[12] = (__bf16)x3.x; f[13] = (__bf16)x3.y; f[14] = (__bf16)x3.z; f[15] = (__bf16)x3.w;
    return f;
}

// ---------------- Weight packer: fp32 (N,K) row-major -> bf16 fragments ----------------
// Fragment id = (tn*kTiles + kt)*32 + lane ; 16 bf16 per fragment (32B, aligned).
__global__ void pack_w_kernel(const float* __restrict__ W, int N, int K,
                              __bf16* __restrict__ P) {
    const int kTiles = K >> 5;
    const int total  = (N >> 4) * kTiles * 32;
    const int idx = blockIdx.x * blockDim.x + threadIdx.x;
    if (idx >= total) return;
    const int lane = idx & 31;
    const int frag = idx >> 5;
    const int kt = frag % kTiles;
    const int tn = frag / kTiles;
    const int n    = (tn << 4) + (lane & 15);
    const int half = lane >> 4;
    __bf16* dst = P + (size_t)idx * 16;
#pragma unroll
    for (int e = 0; e < 16; ++e) {
        const int k = (kt << 5) + 16 * (e >> 3) + 8 * half + (e & 7);
        dst[e] = (__bf16)W[(size_t)n * K + k];
    }
}

// ---------------- WMMA GEMM: C[m,n] = sum_k A[m,k]*Wpacked[n,k] (+bias)(+add)(softplus)
// One wave32 computes a 16 x (16*NT) output strip; NT is compile-time so the
// inner loop is a branch-free unrolled WMMA sequence with pinned accumulators.
// Requires M%16==0, (N/16)%NT==0, K%32==0.
template <int NT>
__global__ void gemm_wmma_bf16(const float* __restrict__ A, int lda,
                               const v16bf* __restrict__ Wp,
                               float* __restrict__ C, int ldc,
                               const float* __restrict__ addSrc,
                               const float* __restrict__ bias,
                               int M, int N, int K, int act /*0=none,1=softplus*/) {
    const int lane = threadIdx.x & 31;
    const int wave = threadIdx.x >> 5;
    const int tilesN  = N >> 4;
    const int kTiles  = K >> 5;
    const int nGroups = tilesN / NT;
    const int group = blockIdx.x * 8 + wave;
    if (group >= (M >> 4) * nGroups) return;        // wave-uniform early out
    const int tm  = group / nGroups;
    const int tn0 = (group % nGroups) * NT;
    const int m0 = tm << 4;
    const int r = lane & 15, half = lane >> 4;

    v8f acc[NT];
#pragma unroll
    for (int j = 0; j < NT; ++j) acc[j] = (v8f){0.f,0.f,0.f,0.f,0.f,0.f,0.f,0.f};

    const float* arow  = A + (size_t)(m0 + r) * lda + 8 * half;
    const v16bf* wbase = Wp + (size_t)tn0 * kTiles * 32 + lane;
    for (int ki = 0; ki < kTiles; ++ki) {
        if (ki + 1 < kTiles) __builtin_prefetch(arow + (ki << 5) + 32, 0, 0);
        const v16bf af = load_frag_f32(arow + (ki << 5));
#pragma unroll
        for (int j = 0; j < NT; ++j) {
            const v16bf wf = wbase[((size_t)j * kTiles + ki) * 32];
            acc[j] = __builtin_amdgcn_wmma_f32_16x16x32_bf16(
                false, af, false, wf, (short)0, acc[j], false, false);
        }
    }

    // D layout: lane -> n = lane&15 ; VGPR v -> m = v + 8*(lane>>4)
    const int nlocal = lane & 15;
#pragma unroll
    for (int j = 0; j < NT; ++j) {
        const int n = ((tn0 + j) << 4) + nlocal;
        const float b = bias ? bias[n] : 0.0f;
#pragma unroll
        for (int v = 0; v < 8; ++v) {
            const int m = m0 + v + 8 * half;
            float val = acc[j][v] + b;
            if (addSrc) val += addSrc[(size_t)m * ldc + n];
            if (act == 1) val = (val > 20.f) ? val : log1pf(__expf(val));
            C[(size_t)m * ldc + n] = val;
        }
    }
}

// ---------------- Wcomb[n,k] = sum_{r<8} dtw[n,r] * xpw[r,k]  (dt_proj ∘ x_proj fusion)
__global__ void dt_comb_kernel(const float* __restrict__ dtw,   // (256,8)
                               const float* __restrict__ xpw,   // (40,256)
                               float* __restrict__ Wc) {        // (256,256)
    const int n = blockIdx.x, k = threadIdx.x;
    float acc = 0.0f;
#pragma unroll
    for (int rr = 0; rr < DT_RANK; ++rr)
        acc += dtw[n * DT_RANK + rr] * xpw[rr * D_INNER + k];
    Wc[(size_t)n * D_INNER + k] = acc;
}

// ---------------- RMSNorm: one 128-thread block per row ----------------
__global__ void rmsnorm_kernel(const float* __restrict__ x, const float* __restrict__ w,
                               float* __restrict__ out) {
    __shared__ float red[D_MODEL];
    const int row = blockIdx.x, tid = threadIdx.x;
    const float v = x[(size_t)row * D_MODEL + tid];
    red[tid] = v * v;
    __syncthreads();
    for (int s = D_MODEL / 2; s > 0; s >>= 1) {
        if (tid < s) red[tid] += red[tid + s];
        __syncthreads();
    }
    const float inv = rsqrtf(red[0] * (1.0f / D_MODEL) + EPS);
    out[(size_t)row * D_MODEL + tid] = v * inv * w[tid];
}

// ---------------- Depthwise causal conv (width 4) + SiLU ----------------
__global__ void conv_silu_kernel(const float* __restrict__ xz,   // (M, 512), xb = [:,0:256]
                                 const float* __restrict__ cw,   // (D_INNER, 4)
                                 const float* __restrict__ cb,   // (D_INNER)
                                 float* __restrict__ xc) {       // (M, D_INNER)
    const int idx = blockIdx.x * blockDim.x + threadIdx.x;
    if (idx >= MROWS * D_INNER) return;
    const int d  = idx & (D_INNER - 1);
    const int bt = idx >> 8;                                     // b*SEQ + t
    const int t  = bt & (SEQ - 1);
    float acc = cb[d];
#pragma unroll
    for (int j = 0; j < D_CONV; ++j) {
        const int tt = t - (D_CONV - 1) + j;
        if (tt >= 0)
            acc += cw[d * D_CONV + j] * xz[(size_t)(bt - (D_CONV - 1) + j) * (2 * D_INNER) + d];
    }
    xc[idx] = acc / (1.0f + __expf(-acc));                       // silu
}

// ---------------- Selective scan, fused with z-gating ----------------
// One block per batch element; thread d owns one channel, 16 states in VGPRs.
__global__ void scan_kernel(const float* __restrict__ bc,     // (M, 32): [0:16]=B, [16:32]=C
                            const float* __restrict__ dlt,    // (M, D_INNER) softplus'd delta
                            const float* __restrict__ xc,     // (M, D_INNER) u
                            const float* __restrict__ xz,     // (M, 512), z = [:,256:512]
                            const float* __restrict__ A_log,  // (D_INNER, D_STATE)
                            const float* __restrict__ Dp,     // (D_INNER)
                            float* __restrict__ y) {          // (M, D_INNER)
    const int b = blockIdx.x, d = threadIdx.x;
    __shared__ float sB[D_STATE], sC[D_STATE];
    float A[D_STATE], h[D_STATE];
#pragma unroll
    for (int n = 0; n < D_STATE; ++n) {
        A[n] = -__expf(A_log[d * D_STATE + n]);
        h[n] = 0.0f;
    }
    const float dskip = Dp[d];
    for (int t = 0; t < SEQ; ++t) {
        const size_t base = (size_t)b * SEQ + t;
        if (d < 2 * D_STATE) {
            const float v = bc[base * (2 * D_STATE) + d];
            if (d < D_STATE) sB[d] = v; else sC[d - D_STATE] = v;
        }
        __syncthreads();
        const float dt = dlt[base * D_INNER + d];
        const float u  = xc[base * D_INNER + d];
        const float du = dt * u;
        float acc = 0.0f;
#pragma unroll
        for (int n = 0; n < D_STATE; ++n) {
            const float dA = __expf(dt * A[n]);
            h[n] = dA * h[n] + du * sB[n];
            acc += h[n] * sC[n];
        }
        acc += u * dskip;
        const float z = xz[base * (2 * D_INNER) + D_INNER + d];
        y[base * D_INNER + d] = acc * (z / (1.0f + __expf(-z)));
        __syncthreads();
    }
}

// ---------------- Classifier head on last token ----------------
__global__ void cls_kernel(const float* __restrict__ hn, const float* __restrict__ w,
                           const float* __restrict__ b, float* __restrict__ out) {
    const int i = threadIdx.x;
    if (i >= BATCH * NUM_CLS) return;
    const int bb = i >> 1, c = i & 1;
    const float* row = hn + ((size_t)bb * SEQ + (SEQ - 1)) * D_MODEL;
    float acc = b[c];
#pragma unroll 4
    for (int k = 0; k < D_MODEL; ++k) acc += row[k] * w[c * D_MODEL + k];
    out[bb * NUM_CLS + c] = acc;
}

// ---------------- Host orchestration ----------------
static inline void run_gemm(const float* A, int lda, const float* W, __bf16* pw,
                            float* C, int ldc, const float* add, const float* bias,
                            int M, int N, int K, int act, hipStream_t s) {
    const int nTiles = N >> 4, kTiles = K >> 5;
    const int frags = nTiles * kTiles * 32;
    pack_w_kernel<<<(frags + 255) / 256, 256, 0, s>>>(W, N, K, pw);
    if ((nTiles & 3) == 0) {
        const int waves = (M >> 4) * (nTiles >> 2);
        gemm_wmma_bf16<4><<<(waves + 7) / 8, 256, 0, s>>>(A, lda, (const v16bf*)pw,
                                                          C, ldc, add, bias, M, N, K, act);
    } else {
        const int waves = (M >> 4) * (nTiles >> 1);
        gemm_wmma_bf16<2><<<(waves + 7) / 8, 256, 0, s>>>(A, lda, (const v16bf*)pw,
                                                          C, ldc, add, bias, M, N, K, act);
    }
}

extern "C" void kernel_launch(void* const* d_in, const int* in_sizes, int n_in,
                              void* d_out, int out_size, void* d_ws, size_t ws_size,
                              hipStream_t stream) {
    (void)in_sizes; (void)n_in; (void)out_size; (void)ws_size;
    const float* x      = (const float*)d_in[0];   // (32,1024,64)
    const float* ipw    = (const float*)d_in[1];   // (128,64)
    const float* ipb    = (const float*)d_in[2];   // (128)
    const float* inw    = (const float*)d_in[3];   // (4,512,128)
    const float* cw     = (const float*)d_in[4];   // (4,256,1,4)
    const float* cb     = (const float*)d_in[5];   // (4,256)
    const float* xpw    = (const float*)d_in[6];   // (4,40,256)
    const float* dtw    = (const float*)d_in[7];   // (4,256,8)
    const float* dtb    = (const float*)d_in[8];   // (4,256)
    const float* A_log  = (const float*)d_in[9];   // (4,256,16)
    const float* Dvec   = (const float*)d_in[10];  // (4,256)
    const float* ow     = (const float*)d_in[11];  // (4,128,256)
    const float* nw     = (const float*)d_in[12];  // (4,128)
    const float* nfw    = (const float*)d_in[13];  // (128)
    const float* clsw   = (const float*)d_in[14];  // (2,128)
    const float* clsb   = (const float*)d_in[15];  // (2)
    float* out = (float*)d_out;

    // Workspace layout (float units)
    float* ws  = (float*)d_ws;
    float* h   = ws;                                 // (M,128)
    float* hn  = h   + (size_t)MROWS * D_MODEL;      // (M,128)
    float* xz  = hn  + (size_t)MROWS * D_MODEL;      // (M,512)
    float* xc  = xz  + (size_t)MROWS * 2 * D_INNER;  // (M,256)
    float* bcb = xc  + (size_t)MROWS * D_INNER;      // (M,32)
    float* dlt = bcb + (size_t)MROWS * 2 * D_STATE;  // (M,256)
    float* yb  = dlt + (size_t)MROWS * D_INNER;      // (M,256)
    float* Wc  = yb  + (size_t)MROWS * D_INNER;      // (256,256) fused dt weight
    __bf16* pw = (__bf16*)(Wc + D_INNER * D_INNER);  // packed weights, <=65536 bf16

    // h = x @ ipw.T + ipb        (32768 x 128 x 64)
    run_gemm(x, INPUT_DIM, ipw, pw, h, D_MODEL, nullptr, ipb,
             MROWS, D_MODEL, INPUT_DIM, 0, stream);

    for (int L = 0; L < N_LAYERS; ++L) {
        rmsnorm_kernel<<<MROWS, D_MODEL, 0, stream>>>(h, nw + L * D_MODEL, hn);

        // xz = hn @ in_proj.T    (32768 x 512 x 128)
        run_gemm(hn, D_MODEL, inw + (size_t)L * 2 * D_INNER * D_MODEL, pw,
                 xz, 2 * D_INNER, nullptr, nullptr, MROWS, 2 * D_INNER, D_MODEL, 0, stream);

        // depthwise conv + silu on xb half -> xc
        conv_silu_kernel<<<(MROWS * D_INNER) / 256, 256, 0, stream>>>(
            xz, cw + (size_t)L * D_INNER * D_CONV, cb + (size_t)L * D_INNER, xc);

        // bc = xc @ xpw[8:40].T  (32768 x 32 x 256)
        run_gemm(xc, D_INNER,
                 xpw + (size_t)L * (DT_RANK + 2 * D_STATE) * D_INNER + DT_RANK * D_INNER, pw,
                 bcb, 2 * D_STATE, nullptr, nullptr, MROWS, 2 * D_STATE, D_INNER, 0, stream);

        // Wcomb = dtw @ xpw[:8]  (256 x 256), then delta = softplus(xc @ Wcomb.T + dtb)
        dt_comb_kernel<<<D_INNER, D_INNER, 0, stream>>>(
            dtw + (size_t)L * D_INNER * DT_RANK,
            xpw + (size_t)L * (DT_RANK + 2 * D_STATE) * D_INNER, Wc);
        run_gemm(xc, D_INNER, Wc, pw, dlt, D_INNER, nullptr, dtb + (size_t)L * D_INNER,
                 MROWS, D_INNER, D_INNER, 1, stream);

        // selective scan + z-gate -> yb
        scan_kernel<<<BATCH, D_INNER, 0, stream>>>(
            bcb, dlt, xc, xz, A_log + (size_t)L * D_INNER * D_STATE,
            Dvec + (size_t)L * D_INNER, yb);

        // h = h + yb @ out_proj.T  (32768 x 128 x 256)
        run_gemm(yb, D_INNER, ow + (size_t)L * D_MODEL * D_INNER, pw,
                 h, D_MODEL, h, nullptr, MROWS, D_MODEL, D_INNER, 0, stream);
    }

    rmsnorm_kernel<<<MROWS, D_MODEL, 0, stream>>>(h, nfw, hn);
    cls_kernel<<<1, 64, 0, stream>>>(hn, clsw, clsb, out);
}
